// CausalSelfAttention_52905407152466
// MI455X (gfx1250) — 8-bit matrix operand order repaired and checked by reading, compiled, not run on hardware
//
#include <hip/hip_runtime.h>
#include <cstdint>

typedef __attribute__((ext_vector_type(8))) int   v8i;
typedef __attribute__((ext_vector_type(4))) int   v4i;
typedef __attribute__((ext_vector_type(8))) float v8f;
typedef __attribute__((ext_vector_type(2))) float v2f;

#define BATCH 2
#define SEQ   2048
#define CDIM  1024
#define NHEAD 16
#define HDIM  64
#define ROWS  (BATCH*SEQ)

#if defined(__has_builtin)
#if __has_builtin(__builtin_amdgcn_global_load_async_to_lds_b128)
#define HAVE_ASYNC_LDS 1
#endif
#endif

#if defined(HAVE_ASYNC_LDS)
typedef __attribute__((address_space(1))) v4i g1_v4i;
typedef __attribute__((address_space(3))) v4i l3_v4i;

__device__ __forceinline__ void async_copy16(const float* gsrc, float* ldst) {
  g1_v4i* gp = (g1_v4i*)(uintptr_t)gsrc;
  l3_v4i* lp = (l3_v4i*)(uint32_t)(uintptr_t)ldst;
  __builtin_amdgcn_global_load_async_to_lds_b128(gp, lp, 0, 0);
}

__device__ __forceinline__ void wait_async0() {
#if __has_builtin(__builtin_amdgcn_s_wait_asynccnt)
  __builtin_amdgcn_s_wait_asynccnt(0);
#else
  asm volatile("s_wait_asynccnt 0" ::: "memory");
#endif
}
#endif

__global__ __launch_bounds__(256) void abssum_kernel(const float* __restrict__ W,
                                                     float* __restrict__ part) {
  float s = 0.f;
  for (size_t i = (size_t)blockIdx.x * 256 + threadIdx.x; i < (size_t)CDIM * CDIM;
       i += (size_t)64 * 256)
    s += fabsf(W[i]);
  for (int off = 16; off; off >>= 1) s += __shfl_xor(s, off, 32);
  __shared__ float red[8];
  if ((threadIdx.x & 31) == 0) red[threadIdx.x >> 5] = s;
  __syncthreads();
  if (threadIdx.x == 0) {
    float t = 0.f;
    for (int i = 0; i < 8; ++i) t += red[i];
    part[blockIdx.x] = t;
  }
}

__global__ __launch_bounds__(256) void wquant_kernel(const float* __restrict__ W,
                                                     const float* __restrict__ part,
                                                     float* __restrict__ wscale,
                                                     int8_t* __restrict__ Wq) {
  __shared__ float ssc;
  if (threadIdx.x == 0) {
    float s = 0.f;
    for (int i = 0; i < 64; ++i) s += part[i];
    float sc = fmaxf(s / ((float)CDIM * (float)CDIM), 1e-5f);
    ssc = sc;
    if (blockIdx.x == 0) wscale[0] = sc;
  }
  __syncthreads();
  float inv = 1.0f / ssc;
  for (size_t i = (size_t)blockIdx.x * 256 + threadIdx.x; i < (size_t)CDIM * CDIM;
       i += (size_t)gridDim.x * 256) {
    float q = rintf(W[i] * inv);
    q = fminf(fmaxf(q, -1.f), 1.f);
    Wq[i] = (int8_t)q;
  }
}

__global__ __launch_bounds__(256) void aquant_kernel(const float* __restrict__ X,
                                                     int8_t* __restrict__ Xq,
                                                     float* __restrict__ rs) {
  const int row = blockIdx.x;
  const float* xr = X + (size_t)row * CDIM;
  float mx = 0.f;
  for (int c = threadIdx.x; c < CDIM; c += 256) mx = fmaxf(mx, fabsf(xr[c]));
  for (int off = 16; off; off >>= 1) mx = fmaxf(mx, __shfl_xor(mx, off, 32));
  __shared__ float red[8];
  if ((threadIdx.x & 31) == 0) red[threadIdx.x >> 5] = mx;
  __syncthreads();
  float amax = fmaxf(fmaxf(fmaxf(red[0], red[1]), fmaxf(red[2], red[3])),
                     fmaxf(fmaxf(red[4], red[5]), fmaxf(red[6], red[7])));
  amax = fmaxf(amax, 1e-5f);
  const float qs = 127.0f / amax;
  if (threadIdx.x == 0) rs[row] = amax / 127.0f;
  for (int c = threadIdx.x; c < CDIM; c += 256) {
    float q = rintf(xr[c] * qs);
    q = fminf(fmaxf(q, -128.f), 127.f);
    Xq[(size_t)row * CDIM + c] = (int8_t)q;
  }
}

__global__ __launch_bounds__(256) void gemm_i8_kernel(const int8_t* __restrict__ xq,
                                                      const float* __restrict__ xrs,
                                                      const int8_t* __restrict__ wq,
                                                      const float* __restrict__ wscale,
                                                      float* __restrict__ out) {
  const int lane = threadIdx.x & 31;
  const int wave = threadIdx.x >> 5;
  const int m    = lane & 15;
  const int half = lane >> 4;
  const int rowbase = blockIdx.y * 64 + (wave & 3) * 16;
  const int colbase = blockIdx.x * 64 + (wave >> 2) * 32;

  v8i acc0 = {0, 0, 0, 0, 0, 0, 0, 0};
  v8i acc1 = {0, 0, 0, 0, 0, 0, 0, 0};
  const float wsc = wscale[0];

  for (int kk = 0; kk < CDIM; kk += 64) {
    v8i a;
    const int8_t* arow = xq + (size_t)(rowbase + m) * CDIM + kk + half * 16;
#pragma unroll
    for (int p = 0; p < 4; ++p) {
      int2 t = *(const int2*)(arow + p * 8 + (p >> 1) * 16);
      a[2 * p] = t.x;
      a[2 * p + 1] = t.y;
    }
    v8i b0, b1;
    const int8_t* brow0 = wq + (size_t)(colbase + m) * CDIM + kk + half * 16;
    const int8_t* brow1 = brow0 + (size_t)16 * CDIM;
#pragma unroll
    for (int q = 0; q < 2; ++q) {
      int4 t0 = *(const int4*)(brow0 + q * 32);
      b0[4 * q + 0] = t0.x; b0[4 * q + 1] = t0.y; b0[4 * q + 2] = t0.z; b0[4 * q + 3] = t0.w;
      int4 t1 = *(const int4*)(brow1 + q * 32);
      b1[4 * q + 0] = t1.x; b1[4 * q + 1] = t1.y; b1[4 * q + 2] = t1.z; b1[4 * q + 3] = t1.w;
    }
    if (kk + 64 < CDIM) {
      __builtin_prefetch(arow + 64, 0, 0);
      __builtin_prefetch(brow0 + 64, 0, 0);
      __builtin_prefetch(brow1 + 64, 0, 0);
    }
    acc0 = __builtin_amdgcn_wmma_i32_16x16x64_iu8(true, a, true, b0, acc0, false, false);
    acc1 = __builtin_amdgcn_wmma_i32_16x16x64_iu8(true, a, true, b1, acc1, false, false);
  }

#pragma unroll
  for (int r = 0; r < 8; ++r) {
    const int row = rowbase + r + half * 8;
    const float d = wsc * xrs[row];
    out[(size_t)row * CDIM + colbase + m]      = (float)acc0[r] * d;
    out[(size_t)row * CDIM + colbase + 16 + m] = (float)acc1[r] * d;
  }
}

__global__ __launch_bounds__(256) void rope_kernel(float* __restrict__ X) {
  const int idx = blockIdx.x * 256 + threadIdx.x;
  if (idx >= ROWS * (CDIM / 2)) return;
  const int row = idx >> 9;
  const int pr  = idx & 511;
  const int h   = pr >> 5;
  const int d   = pr & 31;
  const int t   = row & (SEQ - 1);
  const float freq = __powf(10000.0f, -(float)(2 * d) / 64.0f);
  const float ang  = (float)t * freq;
  const float c = __cosf(ang), s = __sinf(ang);
  float* p1 = X + (size_t)row * CDIM + h * HDIM + d;
  float* p2 = p1 + 32;
  const float x1 = *p1, x2 = *p2;
  *p1 = x1 * c - x2 * s;
  *p2 = x1 * s + x2 * c;
}

#define KS_LD 68
#define PS_LD 17

__global__ __launch_bounds__(128) void attn_kernel(const float* __restrict__ Q,
                                                   const float* __restrict__ K,
                                                   const float* __restrict__ V,
                                                   float* __restrict__ Y) {
  __shared__ float Ks[16 * KS_LD];
  __shared__ float Vs[16 * KS_LD];
  __shared__ float Ps[4 * 16 * PS_LD];

  const int lane = threadIdx.x & 31;
  const int wave = threadIdx.x >> 5;
  const int m    = lane & 15;
  const int half = lane >> 4;
  const int b = blockIdx.y >> 4, h = blockIdx.y & 15;
  const int qblock = blockIdx.x * 64;
  const size_t rowb = (size_t)b * SEQ;
  const int hbase = h * HDIM;

  v2f qa[16];
  {
    const float* qp = Q + (rowb + qblock + wave * 16 + m) * CDIM + hbase + half * 2;
#pragma unroll
    for (int c = 0; c < 16; ++c) { qa[c].x = qp[4 * c]; qa[c].y = qp[4 * c + 1]; }
  }

  v8f o0 = {0,0,0,0,0,0,0,0}, o1 = o0, o2 = o0, o3 = o0;
  float mrow[8], lrow[8];
#pragma unroll
  for (int r = 0; r < 8; ++r) { mrow[r] = -1e30f; lrow[r] = 0.f; }

  const int jend = qblock + 64;
  for (int jb = 0; jb < jend; jb += 16) {
    __syncthreads();
#pragma unroll
    for (int i = 0; i < 2; ++i) {
      const int f = threadIdx.x * 2 + i;
      const int row = f >> 4, c4 = (f & 15) * 4;
      const size_t g = (rowb + jb + row) * CDIM + hbase + c4;
#if defined(HAVE_ASYNC_LDS)
      async_copy16(K + g, &Ks[row * KS_LD + c4]);
      async_copy16(V + g, &Vs[row * KS_LD + c4]);
#else
      *(float4*)&Ks[row * KS_LD + c4] = *(const float4*)(K + g);
      *(float4*)&Vs[row * KS_LD + c4] = *(const float4*)(V + g);
#endif
      if (jb + 16 < jend) {
        __builtin_prefetch(K + g + (size_t)16 * CDIM, 0, 0);
        __builtin_prefetch(V + g + (size_t)16 * CDIM, 0, 0);
      }
    }
#if defined(HAVE_ASYNC_LDS)
    wait_async0();
#endif
    __syncthreads();

    v8f s = {0,0,0,0,0,0,0,0};
#pragma unroll
    for (int c = 0; c < 16; ++c) {
      v2f bk;
      const float* kp = &Ks[m * KS_LD + 4 * c + half * 2];
      bk.x = kp[0]; bk.y = kp[1];
      s = __builtin_amdgcn_wmma_f32_16x16x4_f32(false, qa[c], false, bk,
                                                (short)0, s, false, false);
    }

#pragma unroll
    for (int r = 0; r < 8; ++r) {
      const int qi = qblock + wave * 16 + r + half * 8;
      const int ki = jb + m;
      float sv = (ki <= qi) ? s[r] * 0.125f : -1e30f;
      float mx = sv;
      mx = fmaxf(mx, __shfl_xor(mx, 1, 16));
      mx = fmaxf(mx, __shfl_xor(mx, 2, 16));
      mx = fmaxf(mx, __shfl_xor(mx, 4, 16));
      mx = fmaxf(mx, __shfl_xor(mx, 8, 16));
      const float nm = fmaxf(mrow[r], mx);
      const float p = __expf(sv - nm);
      float ps = p;
      ps += __shfl_xor(ps, 1, 16);
      ps += __shfl_xor(ps, 2, 16);
      ps += __shfl_xor(ps, 4, 16);
      ps += __shfl_xor(ps, 8, 16);
      const float alpha = __expf(mrow[r] - nm);
      lrow[r] = lrow[r] * alpha + ps;
      mrow[r] = nm;
      o0[r] *= alpha; o1[r] *= alpha; o2[r] *= alpha; o3[r] *= alpha;
      Ps[wave * 16 * PS_LD + (r + half * 8) * PS_LD + m] = p;
    }

#pragma unroll
    for (int c = 0; c < 4; ++c) {
      v2f pa;
      const float* pp = &Ps[wave * 16 * PS_LD + m * PS_LD + 4 * c + half * 2];
      pa.x = pp[0]; pa.y = pp[1];
      v2f vb;
      vb.x = Vs[(4 * c + half * 2) * KS_LD + 0 * 16 + m];
      vb.y = Vs[(4 * c + half * 2 + 1) * KS_LD + 0 * 16 + m];
      o0 = __builtin_amdgcn_wmma_f32_16x16x4_f32(false, pa, false, vb, (short)0, o0, false, false);
      vb.x = Vs[(4 * c + half * 2) * KS_LD + 1 * 16 + m];
      vb.y = Vs[(4 * c + half * 2 + 1) * KS_LD + 1 * 16 + m];
      o1 = __builtin_amdgcn_wmma_f32_16x16x4_f32(false, pa, false, vb, (short)0, o1, false, false);
      vb.x = Vs[(4 * c + half * 2) * KS_LD + 2 * 16 + m];
      vb.y = Vs[(4 * c + half * 2 + 1) * KS_LD + 2 * 16 + m];
      o2 = __builtin_amdgcn_wmma_f32_16x16x4_f32(false, pa, false, vb, (short)0, o2, false, false);
      vb.x = Vs[(4 * c + half * 2) * KS_LD + 3 * 16 + m];
      vb.y = Vs[(4 * c + half * 2 + 1) * KS_LD + 3 * 16 + m];
      o3 = __builtin_amdgcn_wmma_f32_16x16x4_f32(false, pa, false, vb, (short)0, o3, false, false);
    }
  }

#pragma unroll
  for (int r = 0; r < 8; ++r) {
    const float inv = 1.0f / lrow[r];
    const int qi = qblock + wave * 16 + r + half * 8;
    float* yp = Y + (rowb + qi) * CDIM + hbase;
    yp[0 * 16 + m] = o0[r] * inv;
    yp[1 * 16 + m] = o1[r] * inv;
    yp[2 * 16 + m] = o2[r] * inv;
    yp[3 * 16 + m] = o3[r] * inv;
  }
}

extern "C" void kernel_launch(void* const* d_in, const int* in_sizes, int n_in,
                              void* d_out, int out_size, void* d_ws, size_t ws_size,
                              hipStream_t stream) {
  (void)in_sizes; (void)n_in; (void)out_size; (void)ws_size;
  const float* x = (const float*)d_in[0];
  const float* Wm[4] = {(const float*)d_in[1], (const float*)d_in[2],
                        (const float*)d_in[3], (const float*)d_in[4]};
  float* out = (float*)d_out;

  char* ws = (char*)d_ws;
  size_t off = 0;
  auto alloc = [&](size_t bytes) -> void* {
    void* p = ws + off;
    off = (off + bytes + 255) & ~(size_t)255;
    return p;
  };
  float*  part = (float*)alloc(4 * 64 * sizeof(float));
  float*  wsc  = (float*)alloc(4 * sizeof(float));
  int8_t* wq8  = (int8_t*)alloc((size_t)4 * CDIM * CDIM);
  int8_t* xq8  = (int8_t*)alloc((size_t)ROWS * CDIM);
  float*  xrs  = (float*)alloc(ROWS * sizeof(float));
  float*  Qb   = (float*)alloc((size_t)ROWS * CDIM * sizeof(float));
  float*  Kb   = (float*)alloc((size_t)ROWS * CDIM * sizeof(float));
  float*  Vb   = (float*)alloc((size_t)ROWS * CDIM * sizeof(float));
  float*  Yb   = (float*)alloc((size_t)ROWS * CDIM * sizeof(float));
  int8_t* yq8  = (int8_t*)alloc((size_t)ROWS * CDIM);
  float*  yrs  = (float*)alloc(ROWS * sizeof(float));

  for (int w = 0; w < 4; ++w)
    abssum_kernel<<<64, 256, 0, stream>>>(Wm[w], part + w * 64);
  for (int w = 0; w < 4; ++w)
    wquant_kernel<<<256, 256, 0, stream>>>(Wm[w], part + w * 64, wsc + w,
                                           wq8 + (size_t)w * CDIM * CDIM);

  aquant_kernel<<<ROWS, 256, 0, stream>>>(x, xq8, xrs);

  const dim3 gg(CDIM / 64, ROWS / 64);
  gemm_i8_kernel<<<gg, 256, 0, stream>>>(xq8, xrs, wq8 + 0 * (size_t)CDIM * CDIM, wsc + 0, Qb);
  gemm_i8_kernel<<<gg, 256, 0, stream>>>(xq8, xrs, wq8 + 1 * (size_t)CDIM * CDIM, wsc + 1, Kb);
  gemm_i8_kernel<<<gg, 256, 0, stream>>>(xq8, xrs, wq8 + 2 * (size_t)CDIM * CDIM, wsc + 2, Vb);

  const int npair = ROWS * (CDIM / 2);
  rope_kernel<<<(npair + 255) / 256, 256, 0, stream>>>(Qb);
  rope_kernel<<<(npair + 255) / 256, 256, 0, stream>>>(Kb);

  attn_kernel<<<dim3(SEQ / 64, BATCH * NHEAD), 128, 0, stream>>>(Qb, Kb, Vb, Yb);

  aquant_kernel<<<ROWS, 256, 0, stream>>>(Yb, yq8, yrs);
  gemm_i8_kernel<<<gg, 256, 0, stream>>>(yq8, yrs, wq8 + 3 * (size_t)CDIM * CDIM, wsc + 3, out);
}
